// MotionInteractionTransformer_32942399161119
// MI455X (gfx1250) — matrix operand order repaired and checked by reading, not compiled, not run on hardware
//
#include <hip/hip_runtime.h>

typedef __attribute__((ext_vector_type(8)))  float  v8f;
typedef __attribute__((ext_vector_type(16))) __bf16 v16bf;

#define DEV static __device__ __forceinline__

// ---------------------------------------------------------------- problem dims
constexpr int   Bb  = 32;              // 2*B
constexpr int   Tt  = 2048;
constexpr int   Dd  = 256;
constexpr int   Hh  = 8;
constexpr int   Ll  = 768;
constexpr int   N77 = 77;
constexpr int   FFc = 1024;
constexpr long  Mm  = (long)Bb * Tt;   // 65536 rows of the main stream
constexpr int   Mxf = Bb * N77;        // 2464 rows of text stream
constexpr float NEGC = -1000000.0f;

// ---------------------------------------------------------------- WMMA helpers
DEV v8f wmma_bf16(v16bf a, v16bf b, v8f c) {
  return __builtin_amdgcn_wmma_f32_16x16x32_bf16(false, a, false, b, (short)0, c,
                                                 false, false);
}

// A-matrix 16x32 bf16 fragment, LDS row-major [m][k], stride in elements.
// Lane l: M = l&15.  Element s(0..15): K = kh*16+s.
DEV v16bf frag_a16x32(const __bf16* p, int stride) {
  const int lane = threadIdx.x & 31;
  const int m  = lane & 15;
  const int kh = lane >> 4;
  const __bf16* q = p + m * stride;
  v16bf r;
#pragma unroll
  for (int j = 0; j < 8; ++j) {
    r[j]     = q[kh * 16 + j];
    r[8 + j] = q[kh * 16 + 8 + j];
  }
  return r;
}

// B-matrix 32x16 bf16 fragment, LDS col-major [n][k], stride in elements.
// Lane l: N = l&15. Lanes 0-15 hold K=0..15, lanes 16-31 hold K=16..31.
DEV v16bf frag_b32x16(const __bf16* p, int stride) {
  const int lane = threadIdx.x & 31;
  const int n  = lane & 15;
  const int kh = lane >> 4;
  const __bf16* q = p + n * stride + kh * 16;
  v16bf r;
#pragma unroll
  for (int j = 0; j < 16; ++j) r[j] = q[j];
  return r;
}

// ---------------------------------------------------------------- main GEMM
enum { EPI_NONE = 0, EPI_KMASK = 1, EPI_VMASK = 2, EPI_RES = 3, EPI_GELU = 4 };

// C[M,N] = epi( A_bf16[M,K] @ W_f32->bf16[K,N] + bias )
// SWAP: A row r is read from (r +/- M/2)  (concat(x2,x1) view for the ICA block)
template<int EPI, bool SWAP, bool OUTBF>
__global__ __launch_bounds__(256)
void gemm_kernel(const __bf16* __restrict__ A, int lda,
                 const float* __restrict__ W, const float* __restrict__ bias,
                 int Mr, int Nc, int Kd,
                 float* __restrict__ outF, __bf16* __restrict__ outB,
                 const float* __restrict__ mask, const float* __restrict__ resid)
{
  constexpr int TM = 128, TN = 64, TK = 64;
  __shared__ __align__(16) __bf16 sA[TM * TK];   // [m][k]  16 KB
  __shared__ __align__(16) __bf16 sB[TN * TK];   // [n][k]   8 KB

  const int tid  = threadIdx.x;
  const int lane = tid & 31;
  const int wid  = tid >> 5;
  const int wm   = wid >> 1;          // 0..3 -> 32-row strip
  const int wn   = wid & 1;           // 0..1 -> 32-col strip
  const int m0   = blockIdx.y * TM;
  const int n0   = blockIdx.x * TN;
  const int halfM = Mr >> 1;

  v8f c00 = {}, c01 = {}, c10 = {}, c11 = {};

  for (int k0 = 0; k0 < Kd; k0 += TK) {
    // ---- stage A tile: 128x64 bf16 = 1024 uint4 / 256 threads (b128 loads)
#pragma unroll
    for (int i = 0; i < 4; ++i) {
      int idx = tid + i * 256;
      int m = idx >> 3, q4 = idx & 7;          // 8 uint4 per 64-elem row
      int row = m0 + m;
      uint4 wv = {0u, 0u, 0u, 0u};
      if (row < Mr) {
        int ar = row;
        if (SWAP) ar = (row < halfM) ? (row + halfM) : (row - halfM);
        wv = *(const uint4*)(A + (long)ar * lda + k0 + q4 * 8);
      }
      ((uint4*)sA)[m * 8 + q4] = wv;
    }
    // ---- stage B tile: 64x64 f32 = 1024 float4 / 256 threads, bf16 transpose
#pragma unroll
    for (int i = 0; i < 4; ++i) {
      int idx = tid + i * 256;
      int k = idx >> 4, n4 = (idx & 15) * 4;   // 16 float4 per k-row
      float4 wv = *(const float4*)(W + (long)(k0 + k) * Nc + n0 + n4);
      sB[(n4 + 0) * TK + k] = (__bf16)wv.x;
      sB[(n4 + 1) * TK + k] = (__bf16)wv.y;
      sB[(n4 + 2) * TK + k] = (__bf16)wv.z;
      sB[(n4 + 3) * TK + k] = (__bf16)wv.w;
    }

    // ---- speculative prefetch of the next K-tile (global_prefetch_b8):
    // issued before the barrier so the lines stream toward L2/WGP$ while the
    // WMMAs below execute.  Covers A: 128 rows x 128B, W: 64 rows x 256B.
    if (k0 + TK < Kd) {
      {
        int m = tid >> 1;                       // 2 threads per 128B row
        int half = (tid & 1) * 32;              // 64B granules
        int row = m0 + m;
        if (row < Mr) {
          int ar = row;
          if (SWAP) ar = (row < halfM) ? (row + halfM) : (row - halfM);
          __builtin_prefetch(A + (long)ar * lda + (k0 + TK) + half, 0, 1);
        }
      }
      {
        int wk  = tid & 63;                     // next 64 k-rows of W
        int seg = (tid >> 6) * 16;              // 4 x 64B per 256B row
        __builtin_prefetch(W + (long)(k0 + TK + wk) * Nc + n0 + seg, 0, 1);
      }
    }
    __syncthreads();

#pragma unroll
    for (int kk = 0; kk < TK; kk += 32) {
      v16bf a0 = frag_a16x32(sA + (wm * 32 + 0)  * TK + kk, TK);
      v16bf a1 = frag_a16x32(sA + (wm * 32 + 16) * TK + kk, TK);
      v16bf b0 = frag_b32x16(sB + (wn * 32 + 0)  * TK + kk, TK);
      v16bf b1 = frag_b32x16(sB + (wn * 32 + 16) * TK + kk, TK);
      c00 = wmma_bf16(a0, b0, c00);
      c01 = wmma_bf16(a0, b1, c01);
      c10 = wmma_bf16(a1, b0, c10);
      c11 = wmma_bf16(a1, b1, c11);
    }
    __syncthreads();
  }

  // ---- epilogue.  C/D layout: VGPR r -> row = r + 8*(lane>=16), col = lane&15
  const int kh = lane >> 4;
  const int cn = lane & 15;
  v8f acc[2][2] = { { c00, c01 }, { c10, c11 } };
#pragma unroll
  for (int im = 0; im < 2; ++im) {
#pragma unroll
    for (int in_ = 0; in_ < 2; ++in_) {
      int col = n0 + wn * 32 + in_ * 16 + cn;
      if (col >= Nc) continue;
      float bv = bias[col];
#pragma unroll
      for (int r = 0; r < 8; ++r) {
        long row = (long)m0 + wm * 32 + im * 16 + kh * 8 + r;
        if (row >= Mr) continue;
        float vv = acc[im][in_][r] + bv;
        if constexpr (EPI == EPI_KMASK) vv += (1.0f - mask[row]) * NEGC;
        if constexpr (EPI == EPI_VMASK) vv *= mask[row];
        if constexpr (EPI == EPI_RES)   vv += resid[row * Nc + col];
        if constexpr (EPI == EPI_GELU)
          vv = 0.5f * vv * (1.0f + erff(vv * 0.70710678118654752f));
        if constexpr (OUTBF) outB[row * Nc + col] = (__bf16)vv;
        else                 outF[row * Nc + col] = vv;
      }
    }
  }
}

// ---------------------------------------------------------------- LayerNorm
// wave-per-row LN.  STYL: h = LN*(1+scale_b)+shift_b, then SiLU (styl block).
template<bool STYL>
__global__ __launch_bounds__(256)
void ln_kernel(const float* __restrict__ x, const float* __restrict__ g,
               const float* __restrict__ bt, const float* __restrict__ e,
               int Dm, long rows, int rowsPerBatch, __bf16* __restrict__ out)
{
  const int lane = threadIdx.x & 31;
  const int wid  = threadIdx.x >> 5;
  long row = (long)blockIdx.x * 8 + wid;
  if (row >= rows) return;
  const float* xr = x + row * Dm;
  float vals[24];                     // Dm <= 768 -> <=24 per lane
  float s = 0.f, ss = 0.f;
  int cnt = 0;
  for (int c = lane; c < Dm; c += 32) {
    float v = xr[c];
    vals[cnt++] = v; s += v; ss += v * v;
  }
#pragma unroll
  for (int o = 16; o > 0; o >>= 1) {
    s  += __shfl_xor(s,  o, 32);
    ss += __shfl_xor(ss, o, 32);
  }
  float mean = s / Dm;
  float rstd = rsqrtf(ss / Dm - mean * mean + 1e-5f);
  int b = (int)(row / rowsPerBatch);
  cnt = 0;
  for (int c = lane; c < Dm; c += 32) {
    float v = (vals[cnt++] - mean) * rstd * g[c] + bt[c];
    if (STYL) {
      v = v * (1.0f + e[b * 512 + c]) + e[b * 512 + 256 + c];
      v = v / (1.0f + __expf(-v));    // SiLU
    }
    out[row * Dm + c] = (__bf16)v;
  }
}

__global__ __launch_bounds__(256)
void cvt_kernel(const float* __restrict__ in, __bf16* __restrict__ out, long n4) {
  long i = (long)blockIdx.x * 256 + threadIdx.x;   // one float4 per thread
  if (i >= n4) return;
  float4 v = *(const float4*)(in + i * 4);
  __bf16* o = out + i * 4;
  o[0] = (__bf16)v.x; o[1] = (__bf16)v.y; o[2] = (__bf16)v.z; o[3] = (__bf16)v.w;
}

// e[b,:512] = silu(emb[b]) @ W(1024x512) + bias   (scale | shift)
__global__ __launch_bounds__(256)
void emb_gemm_kernel(const float* __restrict__ emb, const float* __restrict__ w,
                     const float* __restrict__ b, float* __restrict__ e)
{
  __shared__ float se[1024];
  const int bi = blockIdx.x;
  for (int i = threadIdx.x; i < 1024; i += 256) {
    float v = emb[(long)bi * 1024 + i];
    se[i] = v / (1.0f + __expf(-v));
  }
  __syncthreads();
  for (int col = threadIdx.x; col < 512; col += 256) {
    float acc = b[col];
    for (int k = 0; k < 1024; ++k) acc += se[k] * w[(long)k * 512 + col];
    e[(long)bi * 512 + col] = acc;
  }
}

// softmax over head-dim (32 contiguous cols), one thread per (row, head)
__global__ __launch_bounds__(256)
void qsoftmax_kernel(const float* __restrict__ t, __bf16* __restrict__ out, long nrh)
{
  long i = (long)blockIdx.x * 256 + threadIdx.x;
  if (i >= nrh) return;
  const float* p = t + i * 32;
  float v[32];
#pragma unroll
  for (int j = 0; j < 8; ++j) {
    float4 f = *(const float4*)(p + j * 4);
    v[j * 4 + 0] = f.x; v[j * 4 + 1] = f.y; v[j * 4 + 2] = f.z; v[j * 4 + 3] = f.w;
  }
  float mx = -3.0e38f;
#pragma unroll
  for (int j = 0; j < 32; ++j) mx = fmaxf(mx, v[j]);
  float s = 0.f;
#pragma unroll
  for (int j = 0; j < 32; ++j) { v[j] = __expf(v[j] - mx); s += v[j]; }
  float inv = 1.0f / s;
  __bf16* o = out + i * 32;
#pragma unroll
  for (int j = 0; j < 32; ++j) o[j] = (__bf16)(v[j] * inv);
}

// softmax over the sequence axis, one block per (col, batch)
__global__ __launch_bounds__(256)
void ksoftmax_kernel(const float* __restrict__ t, __bf16* __restrict__ out, int Tn)
{
  const int col = blockIdx.x;           // 0..255
  const int b   = blockIdx.y;           // 0..31
  const int lane = threadIdx.x & 31, wid = threadIdx.x >> 5;
  __shared__ float red[8];
  const float* base = t + (long)b * Tn * 256 + col;
  float v[8];
  int cnt = 0;
  float mx = -3.0e38f;
  for (int n = threadIdx.x; n < Tn; n += 256) {
    float x = base[(long)n * 256];
    v[cnt++] = x; mx = fmaxf(mx, x);
  }
#pragma unroll
  for (int o = 16; o > 0; o >>= 1) mx = fmaxf(mx, __shfl_xor(mx, o, 32));
  if (lane == 0) red[wid] = mx;
  __syncthreads();
  float gmx = red[0];
#pragma unroll
  for (int w = 1; w < 8; ++w) gmx = fmaxf(gmx, red[w]);
  __syncthreads();
  float s = 0.f;
  for (int i2 = 0; i2 < cnt; ++i2) { v[i2] = __expf(v[i2] - gmx); s += v[i2]; }
#pragma unroll
  for (int o = 16; o > 0; o >>= 1) s += __shfl_xor(s, o, 32);
  if (lane == 0) red[wid] = s;
  __syncthreads();
  float tot = 0.f;
#pragma unroll
  for (int w = 0; w < 8; ++w) tot += red[w];
  float inv = 1.0f / tot;
  cnt = 0;
  for (int n = threadIdx.x; n < Tn; n += 256)
    out[(long)b * Tn * 256 + (long)n * 256 + col] = (__bf16)(v[cnt++] * inv);
}

// att[b,h,d,l] = sum_n k[b,n,h,d] * v[b,n,h,l]     (WMMA over n, 4 waves/block)
__global__ __launch_bounds__(128)
void kv_att_kernel(const __bf16* __restrict__ kk, const __bf16* __restrict__ vv,
                   float* __restrict__ att, int Tn)
{
  const int bh = blockIdx.x, b = bh >> 3, h = bh & 7;
  __shared__ __align__(16) __bf16 sA[32 * 32];    // [d][n]
  __shared__ __align__(16) __bf16 sB[32 * 32];    // [l][n]
  const int tid = threadIdx.x, lane = tid & 31, wid = tid >> 5;
  const int wm = wid >> 1, wn = wid & 1;
  const __bf16* kb = kk + (long)b * Tn * 256 + h * 32;
  const __bf16* vb = vv + (long)b * Tn * 256 + h * 32;
  const int n = tid & 31;            // sequence element this thread stages
  const int d0 = (tid >> 5) * 8;     // 8 contiguous features (one b128 load)
  v8f c = {};
  for (int n0 = 0; n0 < Tn; n0 += 32) {
    uint4 kq = {0u, 0u, 0u, 0u}, vq = {0u, 0u, 0u, 0u};
    if (n0 + n < Tn) {               // zero-pad the n=77 case
      long off = (long)(n0 + n) * 256 + d0;
      kq = *(const uint4*)(kb + off);
      vq = *(const uint4*)(vb + off);
      if (n0 + 32 + n < Tn) {        // prefetch next chunk
        __builtin_prefetch(kb + off + 32 * 256, 0, 1);
        __builtin_prefetch(vb + off + 32 * 256, 0, 1);
      }
    }
    const __bf16* kp = (const __bf16*)&kq;
    const __bf16* vp = (const __bf16*)&vq;
#pragma unroll
    for (int j = 0; j < 8; ++j) {    // LDS transpose: [d][n]
      sA[(d0 + j) * 32 + n] = kp[j];
      sB[(d0 + j) * 32 + n] = vp[j];
    }
    __syncthreads();
    v16bf a  = frag_a16x32(sA + wm * 16 * 32, 32);
    v16bf bb = frag_b32x16(sB + wn * 16 * 32, 32);
    c = wmma_bf16(a, bb, c);
    __syncthreads();
  }
  const int kh = lane >> 4, cn = lane & 15;
#pragma unroll
  for (int r = 0; r < 8; ++r)
    att[((long)bh * 32 + wm * 16 + kh * 8 + r) * 32 + wn * 16 + cn] = c[r];
}

// out[row, h*32+l] = sum_d q[row, h*32+d] * att[b,h,d,l]   (block-diag GEMM)
__global__ __launch_bounds__(256)
void qy_kernel(const __bf16* __restrict__ q, const float* __restrict__ att,
               float* __restrict__ out)
{
  const int h  = blockIdx.x;
  const long m0 = (long)blockIdx.y * 128;
  const int b  = (int)(m0 >> 11);           // / 2048 (T rows per batch)
  __shared__ __align__(16) __bf16 sA[128 * 32];
  __shared__ __align__(16) __bf16 sB[32 * 32];   // [l][d]
  const int tid = threadIdx.x, lane = tid & 31, wid = tid >> 5;
  const __bf16* qb = q + m0 * 256 + h * 32;
#pragma unroll
  for (int i = 0; i < 2; ++i) {             // 512 uint4 / 256 threads
    int idx = tid + i * 256;
    int m = idx >> 2, q4 = idx & 3;         // 4 uint4 per 32-elem row
    ((uint4*)sA)[m * 4 + q4] = *(const uint4*)(qb + (long)m * 256 + q4 * 8);
  }
  const float* ab = att + ((long)b * 8 + h) * 1024;
  for (int i = tid; i < 1024; i += 256) {
    int d = i >> 5, l = i & 31;
    sB[l * 32 + d] = (__bf16)ab[d * 32 + l];
  }
  __syncthreads();
  v16bf a  = frag_a16x32(sA + wid * 16 * 32, 32);
  v16bf b0 = frag_b32x16(sB, 32);
  v16bf b1 = frag_b32x16(sB + 16 * 32, 32);
  v8f c0 = {}, c1 = {};
  c0 = wmma_bf16(a, b0, c0);
  c1 = wmma_bf16(a, b1, c1);
  const int kh = lane >> 4, cn = lane & 15;
  float* ob = out + m0 * 256 + h * 32;
#pragma unroll
  for (int r = 0; r < 8; ++r) {
    long rr = (long)(wid * 16 + kh * 8 + r) * 256;
    ob[rr + cn]      = c0[r];
    ob[rr + 16 + cn] = c1[r];
  }
}

// ---------------------------------------------------------------- host driver
extern "C" void kernel_launch(void* const* d_in, const int* in_sizes, int n_in,
                              void* d_out, int out_size, void* d_ws, size_t ws_size,
                              hipStream_t stream)
{
  (void)in_sizes; (void)n_in; (void)out_size; (void)ws_size;
  const float* x1   = (const float*)d_in[0];
  const float* x2   = (const float*)d_in[1];
  const float* xf   = (const float*)d_in[2];
  const float* emb  = (const float*)d_in[3];
  const float* mask = (const float*)d_in[4];
  // d_in[5] = timesteps: not used by the reference forward pass.

  auto P = [&](int i) { return (const float*)d_in[i]; };
  // params flattened in jax pytree (sorted-key) order: ca, ffn, ica, sa.
  const float *ca_k_b=P(6),  *ca_k_w=P(7),  *ca_ln_b=P(8),  *ca_ln_g=P(9),
              *ca_pe_b=P(10),*ca_pe_w=P(11),*ca_pln_b=P(12),*ca_pln_g=P(13),
              *ca_po_b=P(14),*ca_po_w=P(15),*ca_q_b=P(16), *ca_q_w=P(17),
              *ca_tln_b=P(18),*ca_tln_g=P(19),*ca_v_b=P(20),*ca_v_w=P(21);
  const float *ffn_l1_b=P(22),*ffn_l1_w=P(23),*ffn_l2_b=P(24),*ffn_l2_w=P(25),
              *ffn_pe_b=P(26),*ffn_pe_w=P(27),*ffn_pln_b=P(28),*ffn_pln_g=P(29),
              *ffn_po_b=P(30),*ffn_po_w=P(31);
  const float *ica_k_b=P(32),*ica_k_w=P(33),*ica_ln_b=P(34),*ica_ln_g=P(35),
              *ica_pe_b=P(36),*ica_pe_w=P(37),*ica_pln_b=P(38),*ica_pln_g=P(39),
              *ica_po_b=P(40),*ica_po_w=P(41),*ica_q_b=P(42),*ica_q_w=P(43),
              *ica_v_b=P(44),*ica_v_w=P(45);
  const float *sa_k_b=P(46), *sa_k_w=P(47), *sa_ln_b=P(48), *sa_ln_g=P(49),
              *sa_pe_b=P(50),*sa_pe_w=P(51),*sa_pln_b=P(52),*sa_pln_g=P(53),
              *sa_po_b=P(54),*sa_po_w=P(55),*sa_q_b=P(56), *sa_q_w=P(57),
              *sa_v_b=P(58), *sa_v_w=P(59);

  // ---- workspace carve-out (all 256B-aligned by construction)
  char* ws = (char*)d_ws;
  const size_t SZX = (size_t)Mm * Dd * 4;                  // 64 MiB fp32 stream
  float*  ws_x   = (float*)(ws);                           // residual stream
  float*  ws_tmp = (float*)(ws + SZX);                     // fp32 scratch [M,256]
  __bf16* ws_q   = (__bf16*)(ws + 2 * SZX);                // bf16 [M,256]
  __bf16* ws_k   = (__bf16*)(ws + 2 * SZX + SZX / 2);
  __bf16* ws_v   = (__bf16*)(ws + 2 * SZX + 2 * (SZX / 2));
  __bf16* ws_ab  = (__bf16*)(ws + 2 * SZX + 3 * (SZX / 2));
  __bf16* ws_h   = (__bf16*)(ws + 2 * SZX + 4 * (SZX / 2));   // bf16 [M,1024]
  char*   p2     = ws + 2 * SZX + 4 * (SZX / 2) + (size_t)Mm * FFc * 2;
  __bf16* ws_xf  = (__bf16*)p2;                               // bf16 [2464,768]
  float*  ws_att = (float*)(p2 + (size_t)Mxf * Ll * 2);       // [32,8,32,32] fp32
  float*  ws_e   = (float*)(p2 + (size_t)Mxf * Ll * 2 +
                            (size_t)Bb * Hh * 32 * 32 * 4);   // [32,512] fp32

  const dim3 blk(256);
  const dim3 gemmDD((Dd + 63) / 64, (unsigned)(Mm / 128));      // 4 x 512
  const dim3 gemmXf((Dd + 63) / 64, (Mxf + 127) / 128);         // 4 x 20
  const unsigned lnB  = (unsigned)(Mm / 8);
  const unsigned qsB  = (unsigned)((Mm * Hh + 255) / 256);

  // x = concat(x1, x2)
  hipMemcpyAsync(ws_x, x1, SZX / 2, hipMemcpyDeviceToDevice, stream);
  hipMemcpyAsync((char*)ws_x + SZX / 2, x2, SZX / 2, hipMemcpyDeviceToDevice, stream);

  auto run_styl = [&](const float* pe_w, const float* pe_b,
                      const float* pg, const float* pb,
                      const float* po_w, const float* po_b, float* outFinal) {
    emb_gemm_kernel<<<dim3(Bb), blk, 0, stream>>>(emb, pe_w, pe_b, ws_e);
    ln_kernel<true><<<dim3(lnB), blk, 0, stream>>>(ws_tmp, pg, pb, ws_e,
                                                   Dd, Mm, Tt, ws_ab);
    gemm_kernel<EPI_RES, false, false><<<gemmDD, blk, 0, stream>>>(
        ws_ab, Dd, po_w, po_b, (int)Mm, Dd, Dd, outFinal, nullptr, nullptr, ws_x);
  };

  // ================= SA =================
  ln_kernel<false><<<dim3(lnB), blk, 0, stream>>>(ws_x, sa_ln_g, sa_ln_b, nullptr,
                                                  Dd, Mm, Tt, ws_ab);
  gemm_kernel<EPI_NONE, false, false><<<gemmDD, blk, 0, stream>>>(
      ws_ab, Dd, sa_q_w, sa_q_b, (int)Mm, Dd, Dd, ws_tmp, nullptr, nullptr, nullptr);
  qsoftmax_kernel<<<dim3(qsB), blk, 0, stream>>>(ws_tmp, ws_q, Mm * Hh);
  gemm_kernel<EPI_KMASK, false, false><<<gemmDD, blk, 0, stream>>>(
      ws_ab, Dd, sa_k_w, sa_k_b, (int)Mm, Dd, Dd, ws_tmp, nullptr, mask, nullptr);
  ksoftmax_kernel<<<dim3(256, Bb), blk, 0, stream>>>(ws_tmp, ws_k, Tt);
  gemm_kernel<EPI_VMASK, false, true><<<gemmDD, blk, 0, stream>>>(
      ws_ab, Dd, sa_v_w, sa_v_b, (int)Mm, Dd, Dd, nullptr, ws_v, mask, nullptr);
  kv_att_kernel<<<dim3(Bb * Hh), dim3(128), 0, stream>>>(ws_k, ws_v, ws_att, Tt);
  qy_kernel<<<dim3(Hh, (unsigned)(Mm / 128)), blk, 0, stream>>>(ws_q, ws_att, ws_tmp);
  run_styl(sa_pe_w, sa_pe_b, sa_pln_g, sa_pln_b, sa_po_w, sa_po_b, ws_x);

  // ================= CA =================
  ln_kernel<false><<<dim3(lnB), blk, 0, stream>>>(ws_x, ca_ln_g, ca_ln_b, nullptr,
                                                  Dd, Mm, Tt, ws_ab);
  ln_kernel<false><<<dim3((Mxf + 7) / 8), blk, 0, stream>>>(
      xf, ca_tln_g, ca_tln_b, nullptr, Ll, Mxf, N77, ws_xf);
  gemm_kernel<EPI_NONE, false, false><<<gemmDD, blk, 0, stream>>>(
      ws_ab, Dd, ca_q_w, ca_q_b, (int)Mm, Dd, Dd, ws_tmp, nullptr, nullptr, nullptr);
  qsoftmax_kernel<<<dim3(qsB), blk, 0, stream>>>(ws_tmp, ws_q, Mm * Hh);
  gemm_kernel<EPI_NONE, false, false><<<gemmXf, blk, 0, stream>>>(
      ws_xf, Ll, ca_k_w, ca_k_b, Mxf, Dd, Ll, ws_tmp, nullptr, nullptr, nullptr);
  ksoftmax_kernel<<<dim3(256, Bb), blk, 0, stream>>>(ws_tmp, ws_k, N77);
  gemm_kernel<EPI_NONE, false, true><<<gemmXf, blk, 0, stream>>>(
      ws_xf, Ll, ca_v_w, ca_v_b, Mxf, Dd, Ll, nullptr, ws_v, nullptr, nullptr);
  kv_att_kernel<<<dim3(Bb * Hh), dim3(128), 0, stream>>>(ws_k, ws_v, ws_att, N77);
  qy_kernel<<<dim3(Hh, (unsigned)(Mm / 128)), blk, 0, stream>>>(ws_q, ws_att, ws_tmp);
  run_styl(ca_pe_w, ca_pe_b, ca_pln_g, ca_pln_b, ca_po_w, ca_po_b, ws_x);

  // ================= ICA =================
  ln_kernel<false><<<dim3(lnB), blk, 0, stream>>>(ws_x, ica_ln_g, ica_ln_b, nullptr,
                                                  Dd, Mm, Tt, ws_ab);
  gemm_kernel<EPI_NONE, false, false><<<gemmDD, blk, 0, stream>>>(
      ws_ab, Dd, ica_q_w, ica_q_b, (int)Mm, Dd, Dd, ws_tmp, nullptr, nullptr, nullptr);
  qsoftmax_kernel<<<dim3(qsB), blk, 0, stream>>>(ws_tmp, ws_q, Mm * Hh);
  gemm_kernel<EPI_KMASK, true, false><<<gemmDD, blk, 0, stream>>>(   // swapped A rows
      ws_ab, Dd, ica_k_w, ica_k_b, (int)Mm, Dd, Dd, ws_tmp, nullptr, mask, nullptr);
  ksoftmax_kernel<<<dim3(256, Bb), blk, 0, stream>>>(ws_tmp, ws_k, Tt);
  gemm_kernel<EPI_NONE, true, true><<<gemmDD, blk, 0, stream>>>(     // swapped A rows
      ws_ab, Dd, ica_v_w, ica_v_b, (int)Mm, Dd, Dd, nullptr, ws_v, nullptr, nullptr);
  kv_att_kernel<<<dim3(Bb * Hh), dim3(128), 0, stream>>>(ws_k, ws_v, ws_att, Tt);
  qy_kernel<<<dim3(Hh, (unsigned)(Mm / 128)), blk, 0, stream>>>(ws_q, ws_att, ws_tmp);
  run_styl(ica_pe_w, ica_pe_b, ica_pln_g, ica_pln_b, ica_po_w, ica_po_b, ws_x);

  // ================= FFN =================
  cvt_kernel<<<dim3((unsigned)((Mm * Dd / 4 + 255) / 256)), blk, 0, stream>>>(
      ws_x, ws_ab, Mm * Dd / 4);
  gemm_kernel<EPI_GELU, false, true><<<dim3(FFc / 64, (unsigned)(Mm / 128)), blk, 0, stream>>>(
      ws_ab, Dd, ffn_l1_w, ffn_l1_b, (int)Mm, FFc, Dd, nullptr, ws_h, nullptr, nullptr);
  gemm_kernel<EPI_NONE, false, false><<<gemmDD, blk, 0, stream>>>(
      ws_h, FFc, ffn_l2_w, ffn_l2_b, (int)Mm, Dd, FFc, ws_tmp, nullptr, nullptr, nullptr);
  emb_gemm_kernel<<<dim3(Bb), blk, 0, stream>>>(emb, ffn_pe_w, ffn_pe_b, ws_e);
  ln_kernel<true><<<dim3(lnB), blk, 0, stream>>>(ws_tmp, ffn_pln_g, ffn_pln_b, ws_e,
                                                 Dd, Mm, Tt, ws_ab);
  gemm_kernel<EPI_RES, false, false><<<gemmDD, blk, 0, stream>>>(
      ws_ab, Dd, ffn_po_w, ffn_po_b, (int)Mm, Dd, Dd,
      (float*)d_out, nullptr, nullptr, ws_x);
}
